// LSTM_13520557048223
// MI455X (gfx1250) — matrix operand order repaired and checked by reading, not compiled, not run on hardware
//
#include <hip/hip_runtime.h>

typedef __attribute__((ext_vector_type(16))) __bf16 v16bf;
typedef __attribute__((ext_vector_type(8)))  float  v8f;

#define B_     4096
#define SDIM   512
#define ADIM   64
#define HDIM   2048
#define INDIM  577
#define KP     640      // input dim padded to multiple of 32 (20 k-steps)
#define FOURH  8192
#define NB2    80       // head cols padded: 64 probs + 1 value + 15 zero

static __device__ __forceinline__ unsigned short f2bf(float f) {
    unsigned u = __float_as_uint(f);
    unsigned r = u + 0x7fffu + ((u >> 16) & 1u);   // round-to-nearest-even
    return (unsigned short)(r >> 16);
}
static __device__ __forceinline__ float sigmoidf_(float x) {
    return 1.0f / (1.0f + __expf(-x));
}
static __device__ __forceinline__ float tanh_fast(float x) {
    return 1.0f - 2.0f / (__expf(2.0f * x) + 1.0f);
}

// ---------------- prep: pack x = [s | a_prev | r_prev] as bf16, K padded to 640
__global__ void k_build_x(const float* __restrict__ s, const float* __restrict__ a,
                          const float* __restrict__ r, unsigned short* __restrict__ xb) {
    int idx = blockIdx.x * blockDim.x + threadIdx.x;
    if (idx >= B_ * KP) return;
    int b = idx / KP, k = idx - b * KP;
    float v = 0.0f;
    if (k < SDIM)                 v = s[b * SDIM + k];
    else if (k < SDIM + ADIM)     v = a[b * ADIM + (k - SDIM)];
    else if (k == SDIM + ADIM)    v = r[b];
    xb[idx] = f2bf(v);
}

// ---------------- prep: Wx -> fragment-ready bf16 B-matrix layout
// frag element idx = (((kb*512 + j)*32 + lane)*16 + e)
// lane<16: K_local = e,   n = 16j + lane
// lane>=16: K_local = 16+e, n = 16j + lane-16
__global__ void k_build_wx(const float* __restrict__ Wx, unsigned short* __restrict__ fb) {
    int idx = blockIdx.x * blockDim.x + threadIdx.x;
    if (idx >= KP * FOURH) return;
    int e    = idx & 15;
    int lane = (idx >> 4) & 31;
    int j    = (idx >> 9) & 511;          // FOURH/16 = 512
    int kb   = idx >> 18;                 // /(16*32*512)
    int k = 32 * kb + ((lane < 16) ? 0 : 16) + e;
    int n = 16 * j + (lane & 15);
    float v = (k < INDIM) ? Wx[(size_t)k * FOURH + n] : 0.0f;
    fb[idx] = f2bf(v);
}

// ---------------- prep: [Wa | Wv | 0] -> fragment-ready bf16 (2048 x 80)
__global__ void k_build_wav(const float* __restrict__ Wa, const float* __restrict__ Wv,
                            unsigned short* __restrict__ fb) {
    int idx = blockIdx.x * blockDim.x + threadIdx.x;
    if (idx >= HDIM * NB2) return;
    int e    = idx & 15;
    int lane = (idx >> 4) & 31;
    int t    = (idx >> 9) % 5;
    int kb   = idx / (16 * 32 * 5);
    int k = 32 * kb + ((lane < 16) ? 0 : 16) + e;
    int n = 16 * t + (lane & 15);
    float v = 0.0f;
    if (n < 64)       v = Wa[(size_t)k * 64 + n];
    else if (n == 64) v = Wv[k];
    fb[idx] = f2bf(v);
}

// ---------------- zvec[n] = bh[n] + sum_k h[k]*Wh[k, n]   (h, c are vectors!)
__global__ void __launch_bounds__(256) k_zvec(const float* __restrict__ h,
                                              const float* __restrict__ Wh,
                                              const float* __restrict__ bh,
                                              float* __restrict__ zvec) {
    __shared__ float hs[HDIM];
    int n = blockIdx.x * 256 + threadIdx.x;
    for (int k = threadIdx.x; k < HDIM; k += 256) hs[k] = h[k];
    __syncthreads();
    float acc = bh[n];
    for (int k = 0; k < HDIM; ++k) acc = fmaf(hs[k], Wh[(size_t)k * FOURH + n], acc);
    zvec[n] = acc;
}

// ---------------- main fused GEMM + LSTM gating -> h_new (bf16)
// grid (32, 64): 128 batch rows x 32 hidden units per block; 8 waves.
__global__ void __launch_bounds__(256) k_gemm1(const unsigned short* __restrict__ xb,
                                               const unsigned short* __restrict__ fb1,
                                               const float* __restrict__ zvec,
                                               const float* __restrict__ c,
                                               unsigned short* __restrict__ hnewb) {
    __shared__ unsigned short bsh[8 * 32 * 16];   // 8 B-fragments, 8 KB
    const int wave   = threadIdx.x >> 5;
    const int lane   = threadIdx.x & 31;
    const int lanelo = lane & 15;
    const int mb = blockIdx.x * 8 + wave;         // this wave's 16-row tile
    const int hb = blockIdx.y * 32;               // hidden-unit base

    v8f acc[4][2];
    const v8f vzero = {0.f,0.f,0.f,0.f,0.f,0.f,0.f,0.f};
    #pragma unroll
    for (int g = 0; g < 4; ++g)
        #pragma unroll
        for (int t = 0; t < 2; ++t) acc[g][t] = vzero;

    const int m       = 16 * mb + lanelo;
    const int koffsel = (lane < 16) ? 0 : 16;
    // this thread cooperatively stages fragment `wave`, lane `lane`
    const int jmine = (wave >> 1) * 128 + blockIdx.y * 2 + (wave & 1);
    // LDS byte offset of this thread's staging slot (low 32 bits of the
    // generic LDS pointer are the offset within the LDS window)
    const unsigned ldsoff =
        (unsigned)(uintptr_t)&bsh[(size_t)(wave * 32 + lane) * 16];

    for (int kb = 0; kb < KP / 32; ++kb) {
        __syncthreads();
        {   // stage 8 B fragments into LDS via async copy (ASYNCcnt-tracked,
            // bypasses VGPRs). INST_OFFSET applies to both LDS and global addr.
            unsigned long long gaddr =
                (unsigned long long)(const void*)(fb1 +
                    (((size_t)kb * 512 + jmine) * 32 + lane) * 16);
            asm volatile("global_load_async_to_lds_b128 %0, %1, off"
                         :: "v"(ldsoff), "v"(gaddr) : "memory");
            asm volatile("global_load_async_to_lds_b128 %0, %1, off offset:16"
                         :: "v"(ldsoff), "v"(gaddr) : "memory");
        }
        if (kb + 1 < KP / 32)  // pull next k-step's fragments toward near cache
            __builtin_prefetch(fb1 + (((size_t)(kb + 1) * 512 + jmine) * 32 + lane) * 16, 0, 3);
        // A fragment: two b128 loads from row-major bf16 x
        union { uint4 u[2]; v16bf v; } af;
        const unsigned short* ap = xb + (size_t)m * KP + 32 * kb + koffsel;
        af.u[0] = *(const uint4*)ap;
        af.u[1] = *(const uint4*)(ap + 8);
        asm volatile("s_wait_asynccnt 0" ::: "memory");  // my copies landed
        __syncthreads();                                 // everyone's landed
        #pragma unroll
        for (int f = 0; f < 8; ++f) {
            union { uint4 u[2]; v16bf v; } bfv;
            const uint4* ls = (const uint4*)(bsh + (size_t)(f * 32 + lane) * 16);
            bfv.u[0] = ls[0];
            bfv.u[1] = ls[1];
            acc[f >> 1][f & 1] = __builtin_amdgcn_wmma_f32_16x16x32_bf16(
                false, af.v, false, bfv.v, (short)0, acc[f >> 1][f & 1], false, false);
        }
    }

    // epilogue: add broadcast z-vector, LSTM gating, write h_new as bf16
    #pragma unroll
    for (int t = 0; t < 2; ++t) {
        const int nh  = hb + 16 * t + lanelo;
        const float cj  = c[nh];
        const float zvi = zvec[0 * HDIM + nh];
        const float zvf = zvec[1 * HDIM + nh];
        const float zvg = zvec[2 * HDIM + nh];
        const float zvo = zvec[3 * HDIM + nh];
        #pragma unroll
        for (int e = 0; e < 8; ++e) {
            float zi = acc[0][t][e] + zvi;
            float zf = acc[1][t][e] + zvf;
            float zg = acc[2][t][e] + zvg;
            float zo = acc[3][t][e] + zvo;
            float ig = sigmoidf_(zi);
            float fg = sigmoidf_(zf);
            float cn = fg * cj + ig * tanh_fast(zg);
            float og = sigmoidf_(zo);
            float hn = og * tanh_fast(cn);
            int me = 16 * mb + e + ((lane < 16) ? 0 : 8);
            hnewb[(size_t)me * HDIM + nh] = f2bf(hn);
        }
    }
}

// ---------------- heads: probs = softmax(h_new@Wa + ba), v = h_new@Wv + bv
// one wave per 16 batch rows; 5 n-tiles (cols 0..63 probs, col 64 value)
__global__ void __launch_bounds__(32) k_heads(const unsigned short* __restrict__ hnewb,
                                              const unsigned short* __restrict__ fb2,
                                              const float* __restrict__ ba,
                                              const float* __restrict__ bv,
                                              float* __restrict__ out) {
    const int lane   = threadIdx.x & 31;
    const int lanelo = lane & 15;
    const int mb = blockIdx.x;
    v8f acc[5];
    const v8f vzero = {0.f,0.f,0.f,0.f,0.f,0.f,0.f,0.f};
    #pragma unroll
    for (int t = 0; t < 5; ++t) acc[t] = vzero;

    const int m       = 16 * mb + lanelo;
    const int koffsel = (lane < 16) ? 0 : 16;
    for (int kb = 0; kb < HDIM / 32; ++kb) {
        union { uint4 u[2]; v16bf v; } af;
        const unsigned short* ap = hnewb + (size_t)m * HDIM + 32 * kb + koffsel;
        af.u[0] = *(const uint4*)ap;
        af.u[1] = *(const uint4*)(ap + 8);
        #pragma unroll
        for (int t = 0; t < 5; ++t) {
            union { uint4 u[2]; v16bf v; } bfv;
            const uint4* bp = (const uint4*)(fb2 + (((size_t)kb * 5 + t) * 32 + lane) * 16);
            bfv.u[0] = bp[0];
            bfv.u[1] = bp[1];
            acc[t] = __builtin_amdgcn_wmma_f32_16x16x32_bf16(
                false, af.v, false, bfv.v, (short)0, acc[t], false, false);
        }
    }
    // biases
    float bt[5];
    #pragma unroll
    for (int t = 0; t < 4; ++t) bt[t] = ba[16 * t + lanelo];
    bt[4] = (lanelo == 0) ? bv[0] : 0.0f;
    #pragma unroll
    for (int t = 0; t < 5; ++t)
        #pragma unroll
        for (int e = 0; e < 8; ++e) acc[t][e] += bt[t];

    // softmax across 64 action cols (N spans 16 lanes x 4 tiles); lanes 0-15
    // and 16-31 are independent row halves, masks <=8 never cross them.
    #pragma unroll
    for (int e = 0; e < 8; ++e) {
        float mx = fmaxf(fmaxf(acc[0][e], acc[1][e]), fmaxf(acc[2][e], acc[3][e]));
        #pragma unroll
        for (int s = 1; s <= 8; s <<= 1) mx = fmaxf(mx, __shfl_xor(mx, s, 32));
        float e0 = __expf(acc[0][e] - mx);
        float e1 = __expf(acc[1][e] - mx);
        float e2 = __expf(acc[2][e] - mx);
        float e3 = __expf(acc[3][e] - mx);
        float sm = e0 + e1 + e2 + e3;
        #pragma unroll
        for (int s = 1; s <= 8; s <<= 1) sm += __shfl_xor(sm, s, 32);
        float inv = 1.0f / sm;
        int me = 16 * mb + e + ((lane < 16) ? 0 : 8);
        out[(size_t)me * 64 +  0 + lanelo] = e0 * inv;
        out[(size_t)me * 64 + 16 + lanelo] = e1 * inv;
        out[(size_t)me * 64 + 32 + lanelo] = e2 * inv;
        out[(size_t)me * 64 + 48 + lanelo] = e3 * inv;
        if (lanelo == 0) out[(size_t)B_ * 64 + me] = acc[4][e];   // value head
    }
}

extern "C" void kernel_launch(void* const* d_in, const int* in_sizes, int n_in,
                              void* d_out, int out_size, void* d_ws, size_t ws_size,
                              hipStream_t stream) {
    const float* s  = (const float*)d_in[0];
    const float* ap = (const float*)d_in[1];
    const float* rp = (const float*)d_in[2];
    const float* h  = (const float*)d_in[3];
    const float* c  = (const float*)d_in[4];
    const float* Wx = (const float*)d_in[5];
    const float* Wh = (const float*)d_in[6];
    const float* bh = (const float*)d_in[7];
    const float* Wa = (const float*)d_in[8];
    const float* ba = (const float*)d_in[9];
    const float* Wv = (const float*)d_in[10];
    const float* bv = (const float*)d_in[11];
    float* out = (float*)d_out;

    char* w = (char*)d_ws;
    unsigned short* xb    = (unsigned short*)w; w += (size_t)B_ * KP * 2;      // 5.0 MB
    unsigned short* fb1   = (unsigned short*)w; w += (size_t)KP * FOURH * 2;   // 10.0 MB
    unsigned short* fb2   = (unsigned short*)w; w += (size_t)HDIM * NB2 * 2;   // 0.31 MB
    float*          zvec  = (float*)w;          w += (size_t)FOURH * 4;        // 32 KB
    unsigned short* hnewb = (unsigned short*)w;                                 // 16 MB

    k_build_x  <<<(B_ * KP + 255) / 256, 256, 0, stream>>>(s, ap, rp, xb);
    k_build_wx <<<(KP * FOURH + 255) / 256, 256, 0, stream>>>(Wx, fb1);
    k_build_wav<<<(HDIM * NB2 + 255) / 256, 256, 0, stream>>>(Wa, Wv, fb2);
    k_zvec     <<<FOURH / 256, 256, 0, stream>>>(h, Wh, bh, zvec);
    k_gemm1    <<<dim3(B_ / 128, HDIM / 32), 256, 0, stream>>>(xb, fb1, zvec, c, hnewb);
    k_heads    <<<B_ / 16, 32, 0, stream>>>(hnewb, fb2, ba, bv, out);
}
